// ABCConvND_23630910062628
// MI455X (gfx1250) — 8-bit matrix operand order repaired and checked by reading, compiled, not run on hardware
//
#include <hip/hip_runtime.h>

typedef __attribute__((ext_vector_type(16))) int   v16i;
typedef __attribute__((ext_vector_type(8)))  float v8f;

#define HH 112
#define WW 112
#define CINC 128
#define FF 256
#define EE 3
#define HO 110
#define WO 110
#define NCHUNK 9

#define FP8_P1 0x38u
#define FP8_M1 0xB8u

#define XQ_BYTES   (16*112*112*128)
#define S_OFF      XQ_BYTES
#define S_BYTES    (16*112*112*4)
#define BETA_OFF   (S_OFF + S_BYTES)
#define BETA_BYTES (16*110*110*4)
#define WFRAG_OFF  (BETA_OFF + BETA_BYTES)

__global__ void abc_signpack(const float* __restrict__ x,
                             char* __restrict__ xq,
                             float* __restrict__ s) {
    int tid   = blockIdx.x * 256 + threadIdx.x;
    int pixel = tid >> 5;
    int lane  = tid & 31;
    const float4 v = *(const float4*)(x + ((size_t)pixel << 7) + (lane << 2));
    unsigned p =  (v.x >= 0.f ? FP8_P1 : FP8_M1)
               | ((v.y >= 0.f ? FP8_P1 : FP8_M1) << 8)
               | ((v.z >= 0.f ? FP8_P1 : FP8_M1) << 16)
               | ((v.w >= 0.f ? FP8_P1 : FP8_M1) << 24);
    ((unsigned*)xq)[(pixel << 5) + lane] = p;
    float part = fabsf(v.x) + fabsf(v.y) + fabsf(v.z) + fabsf(v.w);
    #pragma unroll
    for (int m = 16; m >= 1; m >>= 1) part += __shfl_xor(part, m, 32);
    if (lane == 0) s[pixel] = part;
}

__global__ void abc_beta(const float* __restrict__ s, float* __restrict__ beta) {
    int idx = blockIdx.x * 256 + threadIdx.x;
    if (idx >= 16*110*110) return;
    int wo = idx % 110; int t = idx / 110;
    int ho = t % 110;   int b = t / 110;
    float acc = 0.f;
    #pragma unroll
    for (int kh = 0; kh < 3; ++kh)
        #pragma unroll
        for (int kw = 0; kw < 3; ++kw)
            acc += s[(b*HH + ho + kh)*WW + wo + kw];
    beta[idx] = acc * (1.0f / 1152.0f);
}

__global__ void abc_wpack(const float* __restrict__ k, char* __restrict__ wfrag) {
    int idx = blockIdx.x * 256 + threadIdx.x;
    if (idx >= EE*16*NCHUNK*32*16) return;
    int i      = idx & 15;
    int lane   = (idx >> 4) & 31;
    int tap    = (idx >> 9) % NCHUNK;
    int rest   = (idx >> 9) / NCHUNK;
    int fslice = rest & 15;
    int e      = rest >> 4;
    int f    = fslice * 16 + (lane & 15);
    int quad = i >> 2;
    int kb   = (i >> 3)*64 + ((i >> 1) & 3)*16 + (i & 1)*4 + (lane >> 4)*8;
    int kh = tap / 3, kw = tap % 3;
    unsigned packed = 0;
    #pragma unroll
    for (int j = 0; j < 4; ++j) {
        int cin = kb + j;
        float kv = k[((((size_t)e*3 + kh)*3 + kw)*CINC + cin)*FF + f];
        packed |= (kv >= 0.f ? FP8_P1 : FP8_M1) << (8*j);
    }
    ((unsigned*)wfrag)[idx] = packed;
}

__global__ void __launch_bounds__(256)
abc_main(const char* __restrict__ xq, const float* __restrict__ beta,
         const v16i* __restrict__ wfrag, const float* __restrict__ alphas,
         float* __restrict__ out) {
    __shared__ int ldsA[2][512];

    const int b       = blockIdx.z;
    const int ho      = blockIdx.y;
    const int wo_tile = blockIdx.x % 7;
    const int f_block = blockIdx.x / 7;
    const int wo0     = wo_tile * 16;
    const int wave    = threadIdx.x >> 5;
    const int lane    = threadIdx.x & 31;
    const int fslice  = f_block * 8 + wave;

    v8f acc0 = {}, acc1 = {}, acc2 = {};

    const int abase = (lane & 15) * 32 + (lane >> 4) * 2;
    const size_t estride = (size_t)16 * NCHUNK * 32;

    auto stage = [&](int tap, int buf) {
        const int row  = threadIdx.x >> 4;
        const int col8 = (threadIdx.x & 15) << 3;
        const int kh = tap / 3, kw = tap % 3;
        const int w  = wo0 + row + kw;
        int2 val; val.x = 0; val.y = 0;
        if (w < WW) {
            size_t off = ((size_t)((b*HH + ho + kh)*WW + w) << 7) + col8;
            val = *(const int2*)(xq + off);
        }
        *(int2*)&ldsA[buf][row*32 + (col8 >> 2)] = val;
    };

    stage(0, 0);
    for (int chunk = 0; chunk < NCHUNK; ++chunk) {
        __syncthreads();
        const int cur = chunk & 1;

        v16i a;
        #pragma unroll
        for (int h = 0; h < 2; ++h) {
            const int o = abase + h*16;
            a[8*h+0] = ldsA[cur][o];      a[8*h+1] = ldsA[cur][o + 1];
            a[8*h+2] = ldsA[cur][o + 4];  a[8*h+3] = ldsA[cur][o + 5];
            a[8*h+4] = ldsA[cur][o + 8];  a[8*h+5] = ldsA[cur][o + 9];
            a[8*h+6] = ldsA[cur][o + 12]; a[8*h+7] = ldsA[cur][o + 13];
        }

        if (chunk + 1 < NCHUNK) stage(chunk + 1, cur ^ 1);

        const v16i* wp = wfrag + (((size_t)fslice * NCHUNK + chunk) * 32 + lane);
        if (chunk + 1 < NCHUNK)
            __builtin_prefetch((const char*)(wp + 32), 0, 1);
        v16i b0 = wp[0];
        v16i b1 = wp[estride];
        v16i b2 = wp[2 * estride];

        acc0 = __builtin_amdgcn_wmma_f32_16x16x128_fp8_fp8(a, b0, (short)0, acc0, false, false);
        acc1 = __builtin_amdgcn_wmma_f32_16x16x128_fp8_fp8(a, b1, (short)0, acc1, false, false);
        acc2 = __builtin_amdgcn_wmma_f32_16x16x128_fp8_fp8(a, b2, (short)0, acc2, false, false);
    }

    const int f   = fslice * 16 + (lane & 15);
    const float al0 = alphas[f];
    const float al1 = alphas[FF + f];
    const float al2 = alphas[2*FF + f];
    const int half = lane >> 4;
    const int pixbase = (b*HO + ho)*WO;
    #pragma unroll
    for (int v = 0; v < 8; ++v) {
        int wo = wo0 + half*8 + v;
        if (wo < WO) {
            float z = al0 * acc0[v] + al1 * acc1[v] + al2 * acc2[v];
            out[((size_t)(pixbase + wo) << 8) + f] = beta[pixbase + wo] * z;
        }
    }
}

extern "C" void kernel_launch(void* const* d_in, const int* in_sizes, int n_in,
                              void* d_out, int out_size, void* d_ws, size_t ws_size,
                              hipStream_t stream) {
    const float* x       = (const float*)d_in[0];
    const float* kernels = (const float*)d_in[1];
    const float* alphas  = (const float*)d_in[2];
    float* out = (float*)d_out;

    char*  ws    = (char*)d_ws;
    char*  xq    = ws;
    float* s     = (float*)(ws + S_OFF);
    float* betap = (float*)(ws + BETA_OFF);
    char*  wfrag = ws + WFRAG_OFF;

    abc_signpack<<<25088, 256, 0, stream>>>(x, xq, s);
    abc_beta<<<(16*110*110 + 255) / 256, 256, 0, stream>>>(s, betap);
    abc_wpack<<<(EE*16*NCHUNK*32*16) / 256, 256, 0, stream>>>(kernels, wfrag);

    dim3 grid(14, 110, 16);
    abc_main<<<grid, 256, 0, stream>>>(xq, betap, (const v16i*)wfrag, alphas, out);

    (void)in_sizes; (void)n_in; (void)out_size; (void)ws_size;
}
